// BrokenModel_10496900071486
// MI455X (gfx1250) — matrix operand order repaired and checked by reading, not compiled, not run on hardware
//
#include <hip/hip_runtime.h>
#include <hip/hip_bf16.h>

typedef __bf16 bf16;
typedef __attribute__((ext_vector_type(16))) __bf16 v16bf;
typedef __attribute__((ext_vector_type(8)))  __bf16 v8bf;
typedef __attribute__((ext_vector_type(8)))  float  v8f;

#define B_DIM 256
#define T_DIM 512
#define H_DIM 1024
#define FOURH 4096

// ---------------------------------------------------------------------------
// Kernel 0a: convert W_hh fp32 [4H, H] -> bf16 (row-major, same layout)
// ---------------------------------------------------------------------------
__global__ __launch_bounds__(256) void convert_w_kernel(
    const float* __restrict__ W, bf16* __restrict__ Wb, int n)
{
    int i = blockIdx.x * blockDim.x + threadIdx.x;
    if (i < n) Wb[i] = (bf16)W[i];
}

// ---------------------------------------------------------------------------
// Kernel 0b: zero h (both ping-pong buffers) and c
// ---------------------------------------------------------------------------
__global__ __launch_bounds__(256) void init_state_kernel(
    bf16* __restrict__ h0, bf16* __restrict__ h1, float* __restrict__ c, int n)
{
    int i = blockIdx.x * blockDim.x + threadIdx.x;
    if (i < n) {
        h0[i] = (bf16)0.0f;
        h1[i] = (bf16)0.0f;
        c[i]  = 0.0f;
    }
}

__device__ __forceinline__ float sigmoidf_dev(float x)
{
    return 1.0f / (1.0f + __expf(-x));
}

// Async DMA: 16 bytes global -> LDS, tracked by ASYNCcnt (CDNA5).
// lds_off is the wave-relative LDS byte address (low 32 bits of a flat
// pointer to __shared__), gptr a 64-bit global address.
__device__ __forceinline__ void async_stage16(unsigned lds_off, const bf16* gptr)
{
    unsigned long long ga = (unsigned long long)gptr;
    asm volatile("global_load_async_to_lds_b128 %0, %1, off"
                 :: "v"(lds_off), "v"(ga) : "memory");
}

__device__ __forceinline__ void wait_asynccnt0()
{
    asm volatile("s_wait_asynccnt 0x0" ::: "memory");
}

// ---------------------------------------------------------------------------
// Kernel 1: one LSTM timestep (+ fused y_{t-1} output projection).
//
// Block: 256 threads = 8 waves; tile = 64 batch-rows x 64 hidden-cols x 4
// gates. Wave (mtp, nt): nt = wave&3 selects a 16-col slice, mtp = wave>>2
// selects a 32-row group; each wave owns TWO 16-row M-subtiles so every B
// fragment (from L2-resident bf16 W) feeds two WMMAs.
//
// The 64x64 h tile is double-buffered in LDS and staged with
// GLOBAL_LOAD_ASYNC_TO_LDS_B128 (ASYNCcnt), so the next tile's global->LDS
// DMA runs concurrently with the current tile's 16 WMMAs; loop body is
// branch-free (last K-iteration peeled) with one barrier per iteration.
// ---------------------------------------------------------------------------
__global__ __launch_bounds__(256) void lstm_step_kernel(
    const bf16*  __restrict__ h_in,     // [B, H] bf16 (h_{t-1})
    bf16*        __restrict__ h_out,    // [B, H] bf16 (h_t)
    float*       __restrict__ c_state,  // [B, H] f32 (in-place update)
    const bf16*  __restrict__ Wb,       // [4H, H] bf16 row-major
    const float* __restrict__ x_all,    // inputs [B, T, 1] f32
    const float* __restrict__ W_ih,     // [4H] f32
    const float* __restrict__ bias,     // [4H] f32
    const float* __restrict__ W_out,    // [H] f32
    const float* __restrict__ b_out,    // [1] f32
    float*       __restrict__ out,      // [B, T] f32
    int t)
{
    __shared__ __align__(16) bf16 ldsA[2][64][72];  // double-buffered h tile
    __shared__ float yred[64][4];

    const int tid  = threadIdx.x;
    const int lane = tid & 31;
    const int wave = tid >> 5;
    const int nt   = wave & 3;    // 0..3 : 16-col slice
    const int mtp  = wave >> 2;   // 0..1 : 32-row group
    const int ln   = lane & 15;   // N index within tile / M index for A
    const int hi   = lane >> 4;   // half-wave selector

    const int m0 = blockIdx.y * 64;   // batch-row base
    const int n0 = blockIdx.x * 64;   // hidden-col base (within [0,H))
    const int n_col = n0 + nt * 16 + ln;

    // staging coordinates (per thread): two 16B chunks per tile
    const int srow = tid >> 3;        // 0..31
    const int scg  = (tid & 7) * 8;   // 0,8,...,56
    const bf16* g0 = h_in + (size_t)(m0 + srow) * H_DIM + scg;
    const bf16* g1 = h_in + (size_t)(m0 + srow + 32) * H_DIM + scg;
    const unsigned l0[2] = {
        (unsigned)(unsigned long long)(const void*)&ldsA[0][srow][scg],
        (unsigned)(unsigned long long)(const void*)&ldsA[1][srow][scg] };
    const unsigned l1[2] = {
        (unsigned)(unsigned long long)(const void*)&ldsA[0][srow + 32][scg],
        (unsigned)(unsigned long long)(const void*)&ldsA[1][srow + 32][scg] };

    // preload per-column gate constants (overlaps with GEMM)
    float wih[4], bb[4];
#pragma unroll
    for (int g = 0; g < 4; ++g) {
        wih[g] = W_ih[g * H_DIM + n_col];
        bb[g]  = bias[g * H_DIM + n_col];
    }

    v8f acc[4][2];
#pragma unroll
    for (int g = 0; g < 4; ++g)
#pragma unroll
        for (int s = 0; s < 2; ++s)
#pragma unroll
            for (int i = 0; i < 8; ++i) acc[g][s][i] = 0.0f;

    // one K-chunk of WMMA work on LDS buffer `buf_` at column base `kk_`
    auto compute_tile = [&](int buf_, int kk_) {
#pragma unroll
        for (int k2 = 0; k2 < 64; k2 += 32) {
            v16bf afrag[2];
#pragma unroll
            for (int s = 0; s < 2; ++s) {
                const int r  = mtp * 32 + s * 16 + ln;
                const int c0 = k2 + hi * 8;
                v8bf lo  = *(const v8bf*)&ldsA[buf_][r][c0];
                v8bf hi8 = *(const v8bf*)&ldsA[buf_][r][c0 + 16];
#pragma unroll
                for (int i = 0; i < 8; ++i) {
                    afrag[s][i] = lo[i]; afrag[s][i + 8] = hi8[i];
                }
            }
#pragma unroll
            for (int g = 0; g < 4; ++g) {
                const bf16* wp = Wb + (size_t)(g * H_DIM + n_col) * H_DIM
                                    + kk_ + k2 + hi * 8;
                __builtin_prefetch((const void*)(wp + 64), 0, 3);
                v8bf b0 = *(const v8bf*)wp;
                v8bf b1 = *(const v8bf*)(wp + 16);
                v16bf bfrag;
#pragma unroll
                for (int i = 0; i < 8; ++i) { bfrag[i] = b0[i]; bfrag[i + 8] = b1[i]; }

#pragma unroll
                for (int s = 0; s < 2; ++s) {
                    acc[g][s] = __builtin_amdgcn_wmma_f32_16x16x32_bf16(
                        false, afrag[s], false, bfrag,
                        (short)0, acc[g][s], false, false);
                }
            }
        }
    };

    // ---- prologue: async-stage tile kk=0 into buffer 0 ----
    async_stage16(l0[0], g0);
    async_stage16(l1[0], g1);
    wait_asynccnt0();
    __syncthreads();

    int buf = 0;
    for (int kk = 0; kk < H_DIM - 64; kk += 64) {
        // async DMA of next tile into the other buffer (runs under the WMMAs)
        async_stage16(l0[buf ^ 1], g0 + kk + 64);
        async_stage16(l1[buf ^ 1], g1 + kk + 64);

        compute_tile(buf, kk);

        wait_asynccnt0();      // my DMA landed in LDS
        __syncthreads();       // everyone's DMA landed
        buf ^= 1;
    }
    compute_tile(buf, H_DIM - 64);   // peeled last K-iteration

    // ---- epilogue: bias + x*W_ih, LSTM cell update, write h/c ----
#pragma unroll
    for (int s = 0; s < 2; ++s) {
#pragma unroll
        for (int r = 0; r < 8; ++r) {
            const int M = hi ? (r + 8) : r;
            const int b = m0 + mtp * 32 + s * 16 + M;
            const float x = x_all[(size_t)b * T_DIM + t];

            float gi = acc[0][s][r] + x * wih[0] + bb[0];
            float gf = acc[1][s][r] + x * wih[1] + bb[1];
            float gg = acc[2][s][r] + x * wih[2] + bb[2];
            float go = acc[3][s][r] + x * wih[3] + bb[3];

            float si = sigmoidf_dev(gi);
            float sf = sigmoidf_dev(gf);
            float so = sigmoidf_dev(go);
            float rg = gg > 0.0f ? gg : 0.0f;

            const size_t idx = (size_t)b * H_DIM + n_col;
            float c_old = c_state[idx];
            float c_new = sf * c_old + si * rg;
            float rc    = c_new > 0.0f ? c_new : 0.0f;
            float h_new = so * rc;

            c_state[idx] = c_new;
            h_out[idx]   = (bf16)h_new;
        }
    }

    // ---- fused output projection: y_{t-1} = h_in . W_out + b_out ----------
    // blockIdx.x==0 blocks (4 of them) cover all 256 batch rows.
    if (blockIdx.x == 0 && t > 0) {
        const int row = tid >> 2;   // 0..63
        const int q   = tid & 3;    // quarter of H
        float s = 0.0f;
        const bf16* hr = h_in + (size_t)(m0 + row) * H_DIM + q * 256;
#pragma unroll 8
        for (int i = 0; i < 256; ++i) s += (float)hr[i] * W_out[q * 256 + i];
        yred[row][q] = s;
        __syncthreads();
        if (q == 0) {
            float y = yred[row][0] + yred[row][1] + yred[row][2] + yred[row][3]
                    + b_out[0];
            out[(size_t)(m0 + row) * T_DIM + (t - 1)] = y;
        }
    }
}

// ---------------------------------------------------------------------------
// Kernel 2: final-step projection y[b] = h[b,:] . W_out + b_out (t = T-1)
// ---------------------------------------------------------------------------
__global__ __launch_bounds__(256) void out_proj_kernel(
    const bf16*  __restrict__ h,      // [B, H] bf16
    const float* __restrict__ W_out,  // [H] f32
    const float* __restrict__ b_out,  // [1] f32
    float*       __restrict__ out,    // [B, T] f32
    int t)
{
    __shared__ float red[256];
    const int b   = blockIdx.x;
    const int tid = threadIdx.x;

    float s = 0.0f;
#pragma unroll
    for (int i = 0; i < 4; ++i) {
        const int n = tid + i * 256;
        s += (float)h[(size_t)b * H_DIM + n] * W_out[n];
    }
    red[tid] = s;
    __syncthreads();
#pragma unroll
    for (int off = 128; off > 0; off >>= 1) {
        if (tid < off) red[tid] += red[tid + off];
        __syncthreads();
    }
    if (tid == 0) out[(size_t)b * T_DIM + t] = red[0] + b_out[0];
}

// ---------------------------------------------------------------------------
extern "C" void kernel_launch(void* const* d_in, const int* in_sizes, int n_in,
                              void* d_out, int out_size, void* d_ws, size_t ws_size,
                              hipStream_t stream)
{
    (void)in_sizes; (void)n_in; (void)out_size; (void)ws_size;

    const float* inputs = (const float*)d_in[0];  // [B, T, 1]
    const float* W_ih   = (const float*)d_in[1];  // [4H, 1]
    const float* W_hh   = (const float*)d_in[2];  // [4H, H]
    const float* bias   = (const float*)d_in[3];  // [4H]
    const float* W_out  = (const float*)d_in[4];  // [1, H]
    const float* b_out  = (const float*)d_in[5];  // [1]
    float* out = (float*)d_out;                   // [B, T, 1]

    // Workspace layout (all offsets 16B-aligned):
    //   Wb : 4H*H bf16  = 8 MB
    //   h0 : B*H bf16   = 512 KB
    //   h1 : B*H bf16   = 512 KB
    //   c  : B*H f32    = 1 MB
    char* ws = (char*)d_ws;
    bf16*  Wb = (bf16*)ws;
    bf16*  h0 = (bf16*)(ws + (size_t)FOURH * H_DIM * sizeof(bf16));
    bf16*  h1 = h0 + (size_t)B_DIM * H_DIM;
    float* c  = (float*)(h1 + (size_t)B_DIM * H_DIM);

    const int wn = FOURH * H_DIM;
    convert_w_kernel<<<(wn + 255) / 256, 256, 0, stream>>>(W_hh, Wb, wn);

    const int sn = B_DIM * H_DIM;
    init_state_kernel<<<(sn + 255) / 256, 256, 0, stream>>>(h0, h1, c, sn);

    dim3 grid(H_DIM / 64, B_DIM / 64);   // 16 x 4 = 64 blocks per step
    bf16* hlast = h1;
    for (int t = 0; t < T_DIM; ++t) {
        const bf16* hin  = (t & 1) ? h1 : h0;
        bf16*       hout = (t & 1) ? h0 : h1;
        lstm_step_kernel<<<grid, 256, 0, stream>>>(
            hin, hout, c, Wb, inputs, W_ih, bias, W_out, b_out, out, t);
        hlast = hout;
    }
    // y for the final step (t = T-1)
    out_proj_kernel<<<B_DIM, 256, 0, stream>>>(hlast, W_out, b_out, out, T_DIM - 1);
}
